// Module7_12266426597604
// MI455X (gfx1250) — matrix operand order repaired and checked by reading, not compiled, not run on hardware
//
#include <hip/hip_runtime.h>
#include <hip/hip_bf16.h>
#include <math.h>

// ---------------------------------------------------------------------------
// CDNA5 (gfx1250) WMMA GEMM + GRU/attention pipeline for the hierarchical
// summarizer forward pass. Wave32.
//   - bulk GEMMs: v_wmma_f32_16x16x32_bf16 (bf16 in, fp32 accumulate)
//   - small precision-sensitive scoring GEMMs: v_wmma_f32_16x16x4_f32
//   - staging: batched b128 global loads (issued before any use -> overlapped
//     latency), packed bf16 cvt, b64 LDS stores, gather/row pointers hoisted
//     out of the K loop, global_prefetch_b8
// ---------------------------------------------------------------------------

typedef float  v2f  __attribute__((ext_vector_type(2)));
typedef float  v4f  __attribute__((ext_vector_type(4)));
typedef float  v8f  __attribute__((ext_vector_type(8)));
typedef __bf16 bf16_t;
typedef __bf16 v4bf  __attribute__((ext_vector_type(4)));
typedef __bf16 v8bf  __attribute__((ext_vector_type(8)));
typedef __bf16 v16bf __attribute__((ext_vector_type(16)));

// ---------------- dims ----------------
#define BB   8      // blogs
#define ND   25     // docs per blog
#define NS   15     // sents per doc
#define TT   30     // words per sent
#define SS   40     // events per blog
#define TE   20     // words per event
#define DD   256    // embed dim
#define HH   256    // hidden
#define H2   512    // 2H
#define H3   768    // 3H
#define POSD 64
#define NSENT (BB*ND*NS)   // 3000
#define NDOC  (BB*ND)      // 200
#define NEV   (BB*SS)      // 320

// ===========================================================================
// bf16 WMMA GEMM:  C[M,N] = act( A[M,K] * B + bias ),  fp32 accumulate.
//   A row-major fp32 (lda), optionally row-gathered from an embedding table:
//       A_row = Abase + gidx[row*gstride + goff] * grow
//   B fp32: bTrans=1 -> B is [N,K] row-major (X @ W^T); 0 -> [K,N] row-major
//   batched over blockIdx.z with element strides sA/sB/sC.
// Block: 128 threads (4 waves). Block tile 64x64; per wave 2x2 16x16 tiles.
// K-chunk 32 -> 4 x v_wmma_f32_16x16x32_bf16 per chunk.
// ===========================================================================

// A fragment (16x32 bf16): lane half h holds K in [8h,8h+8) and [16+8h,16+8h+8)
__device__ __forceinline__ v16bf frag_a(const bf16_t* rowp, int half) {
    v8bf lo = *(const v8bf*)(rowp + 8 * half);
    v8bf hi = *(const v8bf*)(rowp + 16 + 8 * half);
    v16bf r;
#pragma unroll
    for (int e = 0; e < 8; ++e) { r[e] = lo[e]; r[8 + e] = hi[e]; }
    return r;
}
// B fragment (32x16 bf16): lane half h holds K in [8h,8h+8) and [16+8h,16+8h+8)
__device__ __forceinline__ v16bf frag_b(const bf16_t* colp, int half) {
    v8bf lo = *(const v8bf*)(colp + 8 * half);
    v8bf hi = *(const v8bf*)(colp + 16 + 8 * half);
    v16bf r;
#pragma unroll
    for (int e = 0; e < 8; ++e) { r[e] = lo[e]; r[8 + e] = hi[e]; }
    return r;
}
__device__ __forceinline__ v4bf cvt4(v4f v) {
    v4bf r;
#pragma unroll
    for (int e = 0; e < 4; ++e) r[e] = (bf16_t)v[e];
    return r;
}

__global__ __launch_bounds__(128) void wmma_gemm_bf16(
    const float* __restrict__ A, long long lda,
    const float* __restrict__ B, long long ldb, int bTrans,
    float* __restrict__ C, long long ldc,
    const float* __restrict__ bias, int act,
    int M, int N, int K,
    const int* __restrict__ gidx, int gstride, int goff, long long grow,
    long long sA, long long sB, long long sC)
{
    // rows padded to 40 elems (80B): every 16B chunk aligned; 80B stride walks
    // all 16 distinct 16B-bank-groups across 16 lanes -> conflict-free b128.
    __shared__ __align__(16) bf16_t As[64][40];   // [m][k]
    __shared__ __align__(16) bf16_t Bs[64][40];   // [n][k]

    const int t    = threadIdx.x;
    const int wave = t >> 5;
    const int lane = t & 31;
    const int half = lane >> 4;
    const int l16  = lane & 15;
    const int wm   = (wave >> 1) * 32;
    const int wn   = (wave & 1) * 32;
    const int rowBase = blockIdx.y * 64;
    const int colBase = blockIdx.x * 64;

    A += (long long)blockIdx.z * sA;
    B += (long long)blockIdx.z * sB;
    C += (long long)blockIdx.z * sC;

    // ---- hoisted per-thread staging map: 4 fixed rows x one float4 K-slice ----
    const int tm = t >> 3;          // 0..15  (row group)
    const int tk = (t & 7) * 4;     // 0,4,...,28 (K-slice within chunk)
    const float* arow[4];
    bool aok[4];
#pragma unroll
    for (int i = 0; i < 4; ++i) {
        int row = rowBase + tm + 16 * i;
        aok[i] = row < M;
        int rr = aok[i] ? row : 0;
        arow[i] = gidx ? A + (long long)gidx[rr * gstride + goff] * grow
                       : A + (long long)rr * lda;
    }
    const float* brow[4];
    bool bok[4];
    if (bTrans) {
#pragma unroll
        for (int i = 0; i < 4; ++i) {
            int col = colBase + tm + 16 * i;
            bok[i] = col < N;
            brow[i] = B + (long long)(bok[i] ? col : 0) * ldb;
        }
    }
    const int tkB = t >> 4;         // 0..7   ('N' path: k row group)
    const int tnB = (t & 15) * 4;   // 0..60  ('N' path: 4 consecutive cols)

    v8f acc[2][2];
#pragma unroll
    for (int i = 0; i < 2; ++i)
#pragma unroll
        for (int j = 0; j < 2; ++j)
#pragma unroll
            for (int e = 0; e < 8; ++e) acc[i][j][e] = 0.0f;

    for (int k0 = 0; k0 < K; k0 += 32) {
        const int kk = k0 + tk;
        const bool kv = kk < K;         // K is always a multiple of 4
        const bool pf = kk + 32 < K;

        // ===== phase 1: issue ALL global loads (8 x b128), no uses yet =====
        v4f va[4], vb[4];
#pragma unroll
        for (int i = 0; i < 4; ++i) {
            va[i] = (v4f){0.0f, 0.0f, 0.0f, 0.0f};
            if (aok[i] && kv) va[i] = *(const v4f*)(arow[i] + kk);
        }
        if (bTrans) {
#pragma unroll
            for (int i = 0; i < 4; ++i) {
                vb[i] = (v4f){0.0f, 0.0f, 0.0f, 0.0f};
                if (bok[i] && kv) vb[i] = *(const v4f*)(brow[i] + kk);
            }
        } else {
#pragma unroll
            for (int i = 0; i < 4; ++i) {
                int k = tkB + 8 * i;
                int col = colBase + tnB;
                vb[i] = (v4f){0.0f, 0.0f, 0.0f, 0.0f};
                if (col + 3 < N && (k0 + k) < K)
                    vb[i] = *(const v4f*)(B + (long long)(k0 + k) * ldb + col);
            }
        }
        // prefetch next K-slice (independent of the loads above)
        if (pf) {
#pragma unroll
            for (int i = 0; i < 4; ++i)
                if (aok[i]) __builtin_prefetch(arow[i] + kk + 32, 0, 1);
            if (bTrans) {
#pragma unroll
                for (int i = 0; i < 4; ++i)
                    if (bok[i]) __builtin_prefetch(brow[i] + kk + 32, 0, 1);
            }
        }

        // ===== phase 2: convert + LDS stores =====
#pragma unroll
        for (int i = 0; i < 4; ++i)
            *(v4bf*)&As[tm + 16 * i][tk] = cvt4(va[i]);
        if (bTrans) {
#pragma unroll
            for (int i = 0; i < 4; ++i)
                *(v4bf*)&Bs[tm + 16 * i][tk] = cvt4(vb[i]);
        } else {
#pragma unroll
            for (int i = 0; i < 4; ++i) {
                int k = tkB + 8 * i;
                v4bf c = cvt4(vb[i]);
#pragma unroll
                for (int j = 0; j < 4; ++j) Bs[tnB + j][k] = c[j];
            }
        }
        __syncthreads();

        // ---- one K=32 step: 4 WMMAs ----
        v16bf a0 = frag_a(&As[wm + l16][0], half);
        v16bf a1 = frag_a(&As[wm + 16 + l16][0], half);
        v16bf b0 = frag_b(&Bs[wn + l16][0], half);
        v16bf b1 = frag_b(&Bs[wn + 16 + l16][0], half);
        acc[0][0] = __builtin_amdgcn_wmma_f32_16x16x32_bf16(false, a0, false, b0, (short)0, acc[0][0], false, false);
        acc[0][1] = __builtin_amdgcn_wmma_f32_16x16x32_bf16(false, a0, false, b1, (short)0, acc[0][1], false, false);
        acc[1][0] = __builtin_amdgcn_wmma_f32_16x16x32_bf16(false, a1, false, b0, (short)0, acc[1][0], false, false);
        acc[1][1] = __builtin_amdgcn_wmma_f32_16x16x32_bf16(false, a1, false, b1, (short)0, acc[1][1], false, false);
        __syncthreads();
    }

    // ---- epilogue: f32 D layout: VGPR v -> M = v + 8*half, N = l16 ----
#pragma unroll
    for (int ti = 0; ti < 2; ++ti)
#pragma unroll
        for (int tj = 0; tj < 2; ++tj)
#pragma unroll
            for (int v = 0; v < 8; ++v) {
                int row = rowBase + wm + ti * 16 + v + half * 8;
                int col = colBase + wn + tj * 16 + l16;
                if (row < M && col < N) {
                    float x = acc[ti][tj][v];
                    if (bias) x += bias[col];
                    if (act == 1) x = tanhf(x);
                    C[(long long)row * ldc + col] = x;
                }
            }
}

// ===========================================================================
// fp32 WMMA GEMM (v_wmma_f32_16x16x4_f32) — used for the small scoring GEMMs
// ===========================================================================
__global__ __launch_bounds__(128) void wmma_gemm_f32(
    const float* __restrict__ A, long long lda,
    const float* __restrict__ B, long long ldb, int bTrans,
    float* __restrict__ C, long long ldc,
    const float* __restrict__ bias, int act,
    int M, int N, int K,
    long long sA, long long sB, long long sC)
{
    __shared__ float As[64][33];   // [m][k]
    __shared__ float Bs[32][68];   // [k][n]

    const int t    = threadIdx.x;
    const int wave = t >> 5;
    const int lane = t & 31;
    const int half = lane >> 4;
    const int l16  = lane & 15;
    const int wm   = (wave >> 1) * 32;
    const int wn   = (wave & 1) * 32;
    const int rowBase = blockIdx.y * 64;
    const int colBase = blockIdx.x * 64;

    A += (long long)blockIdx.z * sA;
    B += (long long)blockIdx.z * sB;
    C += (long long)blockIdx.z * sC;

    v8f acc[2][2];
#pragma unroll
    for (int i = 0; i < 2; ++i)
#pragma unroll
        for (int j = 0; j < 2; ++j)
#pragma unroll
            for (int e = 0; e < 8; ++e) acc[i][j][e] = 0.0f;

    for (int k0 = 0; k0 < K; k0 += 32) {
#pragma unroll
        for (int i = 0; i < 16; ++i) {
            int e = i * 128 + t;
            int m = e >> 5, k = e & 31;
            int row = rowBase + m;
            float v = 0.0f;
            if (row < M && (k0 + k) < K) v = A[(long long)row * lda + k0 + k];
            As[m][k] = v;
        }
        if (bTrans) {
#pragma unroll
            for (int i = 0; i < 16; ++i) {
                int e = i * 128 + t;
                int n = e >> 5, k = e & 31;
                int col = colBase + n;
                float v = 0.0f;
                if (col < N && (k0 + k) < K) v = B[(long long)col * ldb + (k0 + k)];
                Bs[k][n] = v;
            }
        } else {
#pragma unroll
            for (int i = 0; i < 16; ++i) {
                int e = i * 128 + t;
                int k = e >> 6, n = e & 63;
                int col = colBase + n;
                float v = 0.0f;
                if (col < N && (k0 + k) < K) v = B[(long long)(k0 + k) * ldb + col];
                Bs[k][n] = v;
            }
        }
        __syncthreads();
#pragma unroll
        for (int kk = 0; kk < 32; kk += 4) {
            const int ka = kk + 2 * half;
            v2f a0 = { As[wm + l16     ][ka], As[wm + l16     ][ka + 1] };
            v2f a1 = { As[wm + 16 + l16][ka], As[wm + 16 + l16][ka + 1] };
            v2f b0 = { Bs[ka][wn + l16],      Bs[ka + 1][wn + l16]      };
            v2f b1 = { Bs[ka][wn + 16 + l16], Bs[ka + 1][wn + 16 + l16] };
            acc[0][0] = __builtin_amdgcn_wmma_f32_16x16x4_f32(false, a0, false, b0, (short)0, acc[0][0], false, false);
            acc[0][1] = __builtin_amdgcn_wmma_f32_16x16x4_f32(false, a0, false, b1, (short)0, acc[0][1], false, false);
            acc[1][0] = __builtin_amdgcn_wmma_f32_16x16x4_f32(false, a1, false, b0, (short)0, acc[1][0], false, false);
            acc[1][1] = __builtin_amdgcn_wmma_f32_16x16x4_f32(false, a1, false, b1, (short)0, acc[1][1], false, false);
        }
        __syncthreads();
    }
#pragma unroll
    for (int ti = 0; ti < 2; ++ti)
#pragma unroll
        for (int tj = 0; tj < 2; ++tj)
#pragma unroll
            for (int v = 0; v < 8; ++v) {
                int row = rowBase + wm + ti * 16 + v + half * 8;
                int col = colBase + wn + tj * 16 + l16;
                if (row < M && col < N) {
                    float x = acc[ti][tj][v];
                    if (bias) x += bias[col];
                    if (act == 1) x = tanhf(x);
                    C[(long long)row * ldc + col] = x;
                }
            }
}

// ===========================================================================
// Pointwise / utility kernels
// ===========================================================================
__global__ void fill_f32(float* p, long long n, float v) {
    long long i = (long long)blockIdx.x * blockDim.x + threadIdx.x;
    if (i < n) p[i] = v;
}

__global__ void count_nonzero(const int* __restrict__ ids, int rows, int T, int* __restrict__ lens) {
    int r = blockIdx.x * blockDim.x + threadIdx.x;
    if (r >= rows) return;
    int c = 0;
    for (int t = 0; t < T; ++t) c += (ids[r * T + t] != 0);
    lens[r] = c;
}

__device__ __forceinline__ float sigm(float x) { return 1.0f / (1.0f + expf(-x)); }

// GRU cell + fused masked running max-pool (channel offset selects fwd/bwd half)
__global__ void gru_cell(const float* __restrict__ gi, const float* __restrict__ gh,
                         float* __restrict__ h, float* __restrict__ runmax,
                         const int* __restrict__ lens, int tt, int M, int choff) {
    int i = blockIdx.x * blockDim.x + threadIdx.x;
    if (i >= M * HH) return;
    int row = i >> 8, j = i & (HH - 1);
    const float* gir = gi + (long long)row * H3;
    const float* ghr = gh + (long long)row * H3;
    float r  = sigm(gir[j] + ghr[j]);
    float z  = sigm(gir[HH + j] + ghr[HH + j]);
    float nn = tanhf(gir[2 * HH + j] + r * ghr[2 * HH + j]);
    float hp = (1.0f - z) * nn + z * h[i];
    h[i] = hp;
    if (tt < lens[row]) {
        float* rm = runmax + (long long)row * H2 + choff + j;
        *rm = fmaxf(*rm, hp);
    }
}

__global__ void finalize_max(float* buf, const int* __restrict__ lens, int rows, int C) {
    int i = blockIdx.x * blockDim.x + threadIdx.x;
    if (i >= rows * C) return;
    if (lens[i / C] == 0) buf[i] = 0.0f;
}

// ragged max-pool over the event time axis (tanh already applied in GEMM epilogue)
__global__ void event_pool(const float* __restrict__ evbuf, const int* __restrict__ lens,
                           float* __restrict__ out) {
    int i = blockIdx.x * blockDim.x + threadIdx.x;
    if (i >= NEV * H2) return;
    int e = i / H2, c = i % H2;
    int L = lens[e];
    float m = 0.0f;
    if (L > 0) {
        m = -INFINITY;
        for (int t = 0; t < L; ++t)
            m = fmaxf(m, evbuf[((long long)(e * TE + t)) * H2 + c]);
    }
    out[i] = m;
}

// softmax over S=40 (scalar event_rel_b cancels under softmax; omitted)
__global__ void softmax40(float* buf, int rows) {
    int r = blockIdx.x * blockDim.x + threadIdx.x;
    if (r >= rows) return;
    float* p = buf + (long long)r * SS;
    float e[SS];
    float m = -INFINITY;
    for (int s = 0; s < SS; ++s) m = fmaxf(m, p[s]);
    float sum = 0.0f;
    for (int s = 0; s < SS; ++s) { e[s] = expf(p[s] - m); sum += e[s]; }
    float inv = 1.0f / sum;
    for (int s = 0; s < SS; ++s) p[s] = e[s] * inv;
}

__global__ void concat2(const float* __restrict__ a, const float* __restrict__ b,
                        float* __restrict__ o, int rows, int C) {
    long long i = (long long)blockIdx.x * blockDim.x + threadIdx.x;
    if (i >= (long long)rows * 2 * C) return;
    int row = (int)(i / (2 * C)), c = (int)(i % (2 * C));
    o[i] = (c < C) ? a[(long long)row * C + c] : b[(long long)row * C + (c - C)];
}

// context[b,n] = [blog_vec[b] | doc_vec[b,n]]  -> [200,1024]
__global__ void build_context(const float* __restrict__ blog, const float* __restrict__ docv,
                              float* __restrict__ o) {
    int i = blockIdx.x * blockDim.x + threadIdx.x;
    if (i >= NDOC * 2 * H2) return;
    int row = i >> 10, c = i & 1023;
    int b = row / ND;
    o[i] = (c < H2) ? blog[b * H2 + c] : docv[(long long)row * H2 + (c - H2)];
}

__global__ void doc_scores(const float* __restrict__ docv, const float* __restrict__ vsal,
                           const float* __restrict__ dcw, const float* __restrict__ dpe,
                           const float* __restrict__ dpw, const float* __restrict__ dbias,
                           float* __restrict__ out) {
    int i = blockIdx.x * blockDim.x + threadIdx.x;
    if (i >= NDOC) return;
    int b = i / ND, n = i % ND;
    const float* dv = docv + (long long)i * H2;
    float dc = 0.0f, ds = 0.0f;
    for (int h = 0; h < H2; ++h) { dc += dv[h] * dcw[h]; ds += dv[h] * vsal[b * H2 + h]; }
    float dp = 0.0f;
    for (int k = 0; k < POSD; ++k) dp += dpe[n * POSD + k] * dpw[k];
    out[i] = dc + ds + dp + dbias[0];
}

__global__ void sent_scores(const float* __restrict__ sv, const float* __restrict__ w2,
                            const float* __restrict__ scw, const float* __restrict__ spe,
                            const float* __restrict__ spw, const float* __restrict__ sdlw,
                            const float* __restrict__ sbias, const float* __restrict__ docout,
                            float* __restrict__ out) {
    int i = blockIdx.x * blockDim.x + threadIdx.x;
    if (i >= NSENT) return;
    int b = i / (ND * NS), dn = i / NS, s = i % NS;
    const float* r = sv + (long long)i * H2;
    float sc = 0.0f, ss = 0.0f;
    for (int h = 0; h < H2; ++h) { sc += r[h] * scw[h]; ss += r[h] * w2[(long long)dn * H2 + h]; }
    float sp = 0.0f;
    for (int k = 0; k < POSD; ++k) sp += spe[s * POSD + k] * spw[k];
    out[i] = sc + ss + sp + sdlw[0] * docout[b] + sbias[0];
}

// ===========================================================================
// Host-side orchestration
// ===========================================================================
static inline void gemm(hipStream_t st,
                        const float* A, long long lda,
                        const float* B, long long ldb, int bT,
                        float* C, long long ldc,
                        const float* bias, int act, int M, int N, int K,
                        const int* gidx = nullptr, int gstride = 0, int goff = 0, long long grow = 0,
                        int batch = 1, long long sA = 0, long long sB = 0, long long sC = 0) {
    dim3 g((N + 63) / 64, (M + 63) / 64, batch);
    wmma_gemm_bf16<<<g, 128, 0, st>>>(A, lda, B, ldb, bT, C, ldc, bias, act, M, N, K,
                                      gidx, gstride, goff, grow, sA, sB, sC);
}
static inline void gemm_f32(hipStream_t st,
                            const float* A, long long lda,
                            const float* B, long long ldb, int bT,
                            float* C, long long ldc,
                            const float* bias, int act, int M, int N, int K) {
    dim3 g((N + 63) / 64, (M + 63) / 64, 1);
    wmma_gemm_f32<<<g, 128, 0, st>>>(A, lda, B, ldb, bT, C, ldc, bias, act, M, N, K, 0, 0, 0);
}
static inline void fill(hipStream_t st, float* p, long long n, float v) {
    fill_f32<<<(int)((n + 255) / 256), 256, 0, st>>>(p, n, v);
}

extern "C" void kernel_launch(void* const* d_in, const int* in_sizes, int n_in,
                              void* d_out, int out_size, void* d_ws, size_t ws_size,
                              hipStream_t stream) {
    (void)in_sizes; (void)n_in; (void)out_size; (void)ws_size;
    // ---- inputs (setup_inputs insertion order; params dict flattened in order) ----
    const int*   x         = (const int*)d_in[0];
    const int*   doc_nums  = (const int*)d_in[1];
    const int*   doc_lens  = (const int*)d_in[2];
    const int*   events    = (const int*)d_in[3];
    /* d_in[4] event_weights: unused by the reference forward */
    const float* embed     = (const float*)d_in[5];
    auto F = [&](int i) { return (const float*)d_in[i]; };
    // word_rnn 6..13 / sent_rnn 14..21 / doc_rnn 22..29 : Wih_f,Whh_f,bih_f,bhh_f,Wih_b,Whh_b,bih_b,bhh_b
    const float* event_pre_w    = F(30);
    const float* event_pre_b    = F(31);
    const float* event_rel_w    = F(32);
    /* F(33) event_rel_b: scalar added to all softmax logits -> mathematically a no-op */
    const float* sent_mix_w     = F(34);
    const float* sent_mix_b     = F(35);
    const float* doc_content_w  = F(36);
    const float* doc_salience_w = F(37);
    const float* doc_pos_embed  = F(38);
    const float* doc_pos_w      = F(39);
    const float* doc_bias       = F(40);
    const float* sent_content_w = F(41);
    const float* sent_salience_w= F(42);
    const float* sent_pos_embed = F(43);
    const float* sent_pos_w     = F(44);
    const float* sent_doc_label_w = F(45);
    const float* sent_bias      = F(46);

    // ---- workspace carve-up (floats) ----
    char* wsp = (char*)d_ws;
    auto alloc = [&](long long nf) { float* p = (float*)wsp; wsp += nf * sizeof(float); return p; };
    float* evbuf     = alloc((long long)NEV * TE * H2);   // later reused as concat buffer
    float* event_vec = alloc((long long)NEV * H2);
    float* hbuf      = alloc((long long)NSENT * HH);      // GRU hidden state (all layers)
    float* gibuf     = alloc((long long)NSENT * H3);      // gi   (later: tmp, context2)
    float* ghbuf     = alloc((long long)NSENT * H3);      // gh   (later: att logits)
    float* sent_vec  = alloc((long long)NSENT * H2);      // word-GRU running max -> sent_vec
    float* ctxbuf    = alloc((long long)NSENT * H2);
    float* sent_vec2 = alloc((long long)NSENT * H2);
    float* doc_vec   = alloc((long long)NDOC * H2);
    float* blog_vec  = alloc((long long)BB * H2);
    float* v_sal     = alloc((long long)BB * H2);
    float* w2buf     = alloc((long long)NDOC * H2);
    int* lens_x  = (int*)wsp; wsp += NSENT * sizeof(int);
    int* lens_ev = (int*)wsp;
    float* out = (float*)d_out;

    // ---- ragged lengths ----
    count_nonzero<<<(NSENT + 255) / 256, 256, 0, stream>>>(x, NSENT, TT, lens_x);
    count_nonzero<<<(NEV + 255) / 256, 256, 0, stream>>>(events, NEV, TE, lens_ev);

    // ---- event encoding: tanh(embed[events] @ W^T + b), masked maxpool ----
    gemm(stream, embed, 0, event_pre_w, DD, 1, evbuf, H2, event_pre_b, /*tanh*/1,
         NEV * TE, H2, DD, events, 1, 0, DD);
    event_pool<<<(NEV * H2 + 255) / 256, 256, 0, stream>>>(evbuf, lens_ev, event_vec);

    // ---- word-level biGRU with fused masked maxpool -> sent_vec [3000,512] ----
    fill(stream, sent_vec, (long long)NSENT * H2, -INFINITY);
    for (int d = 0; d < 2; ++d) {
        const float* Wih = F(6 + 4 * d), *Whh = F(7 + 4 * d), *bih = F(8 + 4 * d), *bhh = F(9 + 4 * d);
        fill(stream, hbuf, (long long)NSENT * HH, 0.0f);
        for (int s = 0; s < TT; ++s) {
            int tt = d ? (TT - 1 - s) : s;
            gemm(stream, embed, 0, Wih, DD, 1, gibuf, H3, bih, 0, NSENT, H3, DD, x, TT, tt, DD);
            gemm(stream, hbuf, HH, Whh, HH, 1, ghbuf, H3, bhh, 0, NSENT, H3, HH);
            gru_cell<<<(NSENT * HH + 255) / 256, 256, 0, stream>>>(gibuf, ghbuf, hbuf, sent_vec,
                                                                   lens_x, tt, NSENT, d * HH);
        }
    }
    finalize_max<<<(NSENT * H2 + 255) / 256, 256, 0, stream>>>(sent_vec, lens_x, NSENT, H2);

    // ---- bilinear event attention ----
    float* tmp = gibuf;  // [3000,512]
    gemm(stream, sent_vec, H2, event_rel_w, H2, 0, tmp, H2, nullptr, 0, NSENT, H2, H2);
    float* att = ghbuf;  // [3000,40]
    gemm(stream, tmp, H2, event_vec, H2, 1, att, SS, nullptr, 0, ND * NS, SS, H2,
         nullptr, 0, 0, 0, BB, (long long)ND * NS * H2, (long long)SS * H2, (long long)ND * NS * SS);
    softmax40<<<(NSENT + 255) / 256, 256, 0, stream>>>(att, NSENT);
    gemm(stream, att, SS, event_vec, H2, 0, ctxbuf, H2, nullptr, 0, ND * NS, H2, SS,
         nullptr, 0, 0, 0, BB, (long long)ND * NS * SS, (long long)SS * H2, (long long)ND * NS * H2);
    float* catbuf = evbuf;  // reuse: [3000,1024]
    concat2<<<(int)(((long long)NSENT * 2 * H2 + 255) / 256), 256, 0, stream>>>(sent_vec, ctxbuf, catbuf, NSENT, H2);
    gemm(stream, catbuf, 2 * H2, sent_mix_w, 2 * H2, 1, sent_vec2, H2, sent_mix_b, 0, NSENT, H2, 2 * H2);

    // ---- sentence-level biGRU over docs -> doc_vec [200,512] ----
    fill(stream, doc_vec, (long long)NDOC * H2, -INFINITY);
    for (int d = 0; d < 2; ++d) {
        const float* Wih = F(14 + 4 * d), *Whh = F(15 + 4 * d), *bih = F(16 + 4 * d), *bhh = F(17 + 4 * d);
        fill(stream, hbuf, (long long)NDOC * HH, 0.0f);
        for (int s = 0; s < NS; ++s) {
            int tt = d ? (NS - 1 - s) : s;
            gemm(stream, sent_vec2 + (long long)tt * H2, (long long)NS * H2, Wih, H2, 1, gibuf, H3, bih, 0, NDOC, H3, H2);
            gemm(stream, hbuf, HH, Whh, HH, 1, ghbuf, H3, bhh, 0, NDOC, H3, HH);
            gru_cell<<<(NDOC * HH + 255) / 256, 256, 0, stream>>>(gibuf, ghbuf, hbuf, doc_vec,
                                                                  doc_lens, tt, NDOC, d * HH);
        }
    }
    finalize_max<<<(NDOC * H2 + 255) / 256, 256, 0, stream>>>(doc_vec, doc_lens, NDOC, H2);

    // ---- doc-level biGRU over blogs -> blog_vec [8,512] ----
    fill(stream, blog_vec, (long long)BB * H2, -INFINITY);
    for (int d = 0; d < 2; ++d) {
        const float* Wih = F(22 + 4 * d), *Whh = F(23 + 4 * d), *bih = F(24 + 4 * d), *bhh = F(25 + 4 * d);
        fill(stream, hbuf, (long long)BB * HH, 0.0f);
        for (int s = 0; s < ND; ++s) {
            int tt = d ? (ND - 1 - s) : s;
            gemm(stream, doc_vec + (long long)tt * H2, (long long)ND * H2, Wih, H2, 1, gibuf, H3, bih, 0, BB, H3, H2);
            gemm(stream, hbuf, HH, Whh, HH, 1, ghbuf, H3, bhh, 0, BB, H3, HH);
            gru_cell<<<(BB * HH + 255) / 256, 256, 0, stream>>>(gibuf, ghbuf, hbuf, blog_vec,
                                                                doc_nums, tt, BB, d * HH);
        }
    }
    finalize_max<<<(BB * H2 + 255) / 256, 256, 0, stream>>>(blog_vec, doc_nums, BB, H2);

    // ---- doc scores -> out[0:200]  (fp32 WMMA path for fidelity) ----
    gemm_f32(stream, blog_vec, H2, doc_salience_w, H2, 1, v_sal, H2, nullptr, 0, BB, H2, H2);
    doc_scores<<<(NDOC + 255) / 256, 256, 0, stream>>>(doc_vec, v_sal, doc_content_w,
                                                       doc_pos_embed, doc_pos_w, doc_bias, out);

    // ---- sentence scores -> out[200:3200] ----
    float* context2 = gibuf;  // reuse: [200,1024]
    build_context<<<(NDOC * 2 * H2 + 255) / 256, 256, 0, stream>>>(blog_vec, doc_vec, context2);
    gemm_f32(stream, context2, 2 * H2, sent_salience_w, 2 * H2, 1, w2buf, H2, nullptr, 0, NDOC, H2, 2 * H2);
    sent_scores<<<(NSENT + 255) / 256, 256, 0, stream>>>(sent_vec2, w2buf, sent_content_w,
                                                         sent_pos_embed, sent_pos_w,
                                                         sent_doc_label_w, sent_bias,
                                                         out, out + NDOC);
}
